// GCN3D_37873021616797
// MI455X (gfx1250) — matrix operand order repaired and checked by reading, not compiled, not run on hardware
//
#include <hip/hip_runtime.h>
#include <math.h>

// ---------------------------------------------------------------------------
// GCN3D forward for MI455X (gfx1250, wave32).
// Dense GEMMs -> v_wmma_f32_16x16x32_bf16, each wave computes a 16x64 strip
// (4 accumulators) so each A fragment feeds 4 WMMAs: 2 A + 8 B b128-loads per
// 4 WMMAs per K-step. B matrices pre-transposed to [N, Kpad] bf16 so all
// fragment loads are contiguous 16B/32B.
// ---------------------------------------------------------------------------

typedef __attribute__((ext_vector_type(16))) __bf16 v16bf;
typedef __attribute__((ext_vector_type(8)))  float  v8f;

struct __align__(16) U4 { unsigned int x, y, z, w; };
union FragU { v16bf v; U4 q[2]; };

__device__ __forceinline__ unsigned short f2bf(float f) {
    union { float f; unsigned int u; } cv; cv.f = f;
    unsigned int u = cv.u;
    unsigned int r = u + 0x7FFFu + ((u >> 16) & 1u);   // round-to-nearest-even
    return (unsigned short)(r >> 16);
}

// ---------------- small prep kernels ----------------

// normalize columns of a (3,32) direction matrix
__global__ void k_normcols(const float* __restrict__ D, float* __restrict__ O) {
    int k = threadIdx.x;                 // 0..31
    float a = D[k], b = D[32 + k], c = D[64 + k];
    float n = sqrtf(a * a + b * b + c * c);
    float inv = 1.0f / fmaxf(n, 1e-12f);
    O[k] = a * inv; O[32 + k] = b * inv; O[64 + k] = c * inv;
}

// W[K,N] fp32 -> WT[N,K] bf16
__global__ void k_tcvt(const float* __restrict__ W, unsigned short* __restrict__ WT,
                       int K, int N) {
    int idx = blockIdx.x * blockDim.x + threadIdx.x;
    if (idx >= K * N) return;
    int n = idx / K, k = idx % K;
    WT[idx] = f2bf(W[(size_t)k * N + n]);
}

// S[rsrc, lds] fp32 (leading cols cs valid) -> D[rdst, cd] bf16, zero padded
__global__ void k_cvt(const float* __restrict__ S, int lds, int rsrc,
                      unsigned short* __restrict__ D, int cd, int rdst, int cs) {
    long long idx = (long long)blockIdx.x * blockDim.x + threadIdx.x;
    if (idx >= (long long)rdst * cd) return;
    int r = (int)(idx / cd), c = (int)(idx % cd);
    float v = (r < rsrc && c < cs) ? S[(size_t)r * lds + c] : 0.0f;
    D[idx] = f2bf(v);
}

// ---------------- ball query ----------------
// Emulates: idx = sort(where(d2 > r2, N, arange))[:nsample]; pad with first hit.
__global__ void k_ballq(const float* __restrict__ src, int nsrc,
                        const float* __restrict__ q, int nq, int qstride,
                        float r2, int nsample, int* __restrict__ out, int bs) {
    int t = blockIdx.x * blockDim.x + threadIdx.x;
    if (t >= bs * nq) return;
    int b = t / nq, j = t % nq;
    const float* qp = q + ((size_t)b * qstride + j) * 3;
    float px = qp[0], py = qp[1], pz = qp[2];
    const float* sp = src + (size_t)b * nsrc * 3;
    int* o = out + (size_t)t * nsample;
    int cnt = 0;
    for (int n = 0; n < nsrc && cnt < nsample; ++n) {
        float dx = px - sp[n * 3 + 0];
        float dy = py - sp[n * 3 + 1];
        float dz = pz - sp[n * 3 + 2];
        if (dx * dx + dy * dy + dz * dz <= r2) o[cnt++] = n;
    }
    int f0 = (cnt > 0) ? o[0] : 0;
    for (; cnt < nsample; ++cnt) o[cnt] = f0;
}

// ---------------- surface conv (fm0) ----------------
// block = 256 threads = 8 waves; wave w handles vertex blockIdx.x*8+w, lane = out channel
__global__ void k_conv_surface(const float* __restrict__ verts, int V,
                               const int* __restrict__ nbr,
                               const float* __restrict__ dn,
                               float* __restrict__ out, int ldo) {
    __shared__ float dirs[8][32][3];
    int wid = threadIdx.x >> 5, lane = threadIdx.x & 31;
    int gv = blockIdx.x * 8 + wid;
    int b = gv / V, v = gv % V;
    const float* base = verts + (size_t)b * V * 3;
    float px = base[v * 3], py = base[v * 3 + 1], pz = base[v * 3 + 2];
    int nid = nbr[(size_t)gv * 32 + lane];
    float dx = base[nid * 3] - px, dy = base[nid * 3 + 1] - py, dz = base[nid * 3 + 2] - pz;
    float nn = sqrtf(dx * dx + dy * dy + dz * dz);
    float inv = 1.0f / fmaxf(nn, 1e-12f);
    dirs[wid][lane][0] = dx * inv;
    dirs[wid][lane][1] = dy * inv;
    dirs[wid][lane][2] = dz * inv;
    __syncthreads();
    float d0 = dn[lane], d1 = dn[32 + lane], d2 = dn[64 + lane];
    float acc = 0.0f;                      // theta >= 0 so relu(max)=max
    for (int n = 0; n < 32; ++n) {
        float th = dirs[wid][n][0] * d0 + dirs[wid][n][1] * d1 + dirs[wid][n][2] * d2;
        acc = fmaxf(acc, fmaxf(th, 0.0f));
    }
    out[(size_t)gv * ldo + lane] = acc;
}

// ---------------- conv layer aggregation ----------------
// out[:, coff+k] = relu(center_k + max_n relu(<dir_n, dd_k>) * support[nbr_n, k])
__global__ void k_conv_aggregate(const float* __restrict__ verts, int V,
                                 const int* __restrict__ nbr,
                                 const float* __restrict__ dn,
                                 const float* __restrict__ f,   // [bs*V, 64]
                                 float* __restrict__ out, int ldo, int coff) {
    __shared__ float dirs[8][32][3];
    __shared__ int   snid[8][32];
    int wid = threadIdx.x >> 5, lane = threadIdx.x & 31;
    int gv = blockIdx.x * 8 + wid;
    int b = gv / V, v = gv % V;
    const float* base = verts + (size_t)b * V * 3;
    float px = base[v * 3], py = base[v * 3 + 1], pz = base[v * 3 + 2];
    int nid = nbr[(size_t)gv * 32 + lane];
    snid[wid][lane] = nid;
    float dx = base[nid * 3] - px, dy = base[nid * 3 + 1] - py, dz = base[nid * 3 + 2] - pz;
    float nn = sqrtf(dx * dx + dy * dy + dz * dz);
    float inv = 1.0f / fmaxf(nn, 1e-12f);
    dirs[wid][lane][0] = dx * inv;
    dirs[wid][lane][1] = dy * inv;
    dirs[wid][lane][2] = dz * inv;
    __syncthreads();
    float d0 = dn[lane], d1 = dn[32 + lane], d2 = dn[64 + lane];
    float center = f[(size_t)gv * 64 + lane];
    float acc = -INFINITY;
    for (int n = 0; n < 32; ++n) {
        float th = dirs[wid][n][0] * d0 + dirs[wid][n][1] * d1 + dirs[wid][n][2] * d2;
        th = fmaxf(th, 0.0f);
        float sup = f[((size_t)b * V + snid[wid][n]) * 64 + 32 + lane];
        acc = fmaxf(acc, th * sup);
    }
    out[(size_t)gv * ldo + coff + lane] = fmaxf(center + acc, 0.0f);
}

// ---------------- pool (max over 4 ball neighbors, first Vout vertices kept) ----
__global__ void k_pool(const float* __restrict__ fin, int ldin, int C, int Vin,
                       const int* __restrict__ idx, const float* __restrict__ vin,
                       float* __restrict__ fout, int ldout,
                       float* __restrict__ vout, int Vout) {
    int blk = blockIdx.x;
    int b = blk / Vout, j = blk % Vout;
    const int* ip = idx + (size_t)blk * 4;
    int i0 = ip[0], i1 = ip[1], i2 = ip[2], i3 = ip[3];
    const float* fb = fin + (size_t)b * Vin * ldin;
    for (int c = threadIdx.x; c < C; c += blockDim.x) {
        float m = fb[(size_t)i0 * ldin + c];
        m = fmaxf(m, fb[(size_t)i1 * ldin + c]);
        m = fmaxf(m, fb[(size_t)i2 * ldin + c]);
        m = fmaxf(m, fb[(size_t)i3 * ldin + c]);
        fout[(size_t)blk * ldout + c] = m;
    }
    if (threadIdx.x < 3)
        vout[(size_t)blk * 3 + threadIdx.x] = vin[((size_t)b * Vin + j) * 3 + threadIdx.x];
}

// ---------------- nearest index ----------------
__global__ void k_nearest(const float* __restrict__ q, int V,
                          const float* __restrict__ src, int Vs,
                          int* __restrict__ out, int bs) {
    int t = blockIdx.x * blockDim.x + threadIdx.x;
    if (t >= bs * V) return;
    int b = t / V, v = t % V;
    const float* qp = q + ((size_t)b * V + v) * 3;
    float px = qp[0], py = qp[1], pz = qp[2];
    const float* sp = src + (size_t)b * Vs * 3;
    float best = INFINITY; int bi = 0;
    for (int n = 0; n < Vs; ++n) {
        float dx = px - sp[n * 3], dy = py - sp[n * 3 + 1], dz = pz - sp[n * 3 + 2];
        float d2 = dx * dx + dy * dy + dz * dz;
        if (d2 < best) { best = d2; bi = n; }
    }
    out[t] = bi;
}

// ---------------- global feature (max over stage-3 vertices) ----------------
__global__ void k_fglobal(const float* __restrict__ fmC, float* __restrict__ out) {
    int t = blockIdx.x * 256 + threadIdx.x;   // 8*256 total
    int b = t >> 8, c = t & 255;
    float m = -INFINITY;
    for (int v = 0; v < 128; ++v)
        m = fmaxf(m, fmC[((size_t)b * 128 + v) * 256 + c]);
    out[t] = m;
}

// ---------------- fuse concat -> bf16 [16384, 1440] ----------------
__global__ void k_fuse(const float* __restrict__ fmA, const float* __restrict__ fmB,
                       const float* __restrict__ fmC, const int* __restrict__ n1,
                       const int* __restrict__ n2, const float* __restrict__ fg,
                       const float* __restrict__ onehot, unsigned short* __restrict__ out) {
    int row = blockIdx.x;                      // b*2048 + v
    int b = row >> 11;
    int r1 = b * 512 + n1[row];
    int r2 = b * 128 + n2[row];
    for (int c = threadIdx.x; c < 1440; c += 256) {
        float v;
        if      (c <   32) v = fmA[(size_t)row * 96 + c];
        else if (c <   96) v = fmA[(size_t)row * 96 + (c - 32)];
        else if (c <  192) v = fmA[(size_t)row * 96 + (c - 96)];
        else if (c <  320) v = fmB[(size_t)r1 * 192 + (c - 192)];
        else if (c <  480) v = fmB[(size_t)r1 * 192 + (c - 320)];
        else if (c <  672) v = fmB[(size_t)r1 * 192 + (c - 480)];
        else if (c <  896) v = fmC[(size_t)r2 * 256 + (c - 672)];
        else if (c < 1152) v = fmC[(size_t)r2 * 256 + (c - 896)];
        else if (c < 1408) v = fg[b * 256 + (c - 1152)];
        else if (c < 1424) v = onehot[b * 16 + (c - 1408)];
        else               v = 0.0f;
        out[(size_t)row * 1440 + c] = f2bf(v);
    }
}

// ---------------- WMMA bf16 GEMM, 16x64 strip per wave ----------------
// C[M,N] = A[M,K] * BT[N,K]^T (+bias) (+relu).
// Requires: K % 32 == 0, M % 16 == 0, N % 64 == 0 relative to the launched
// wave count (every active wave owns exactly 4 n-tiles).
__global__ __launch_bounds__(128)
void k_gemm(const unsigned short* __restrict__ A, int lda,
            const unsigned short* __restrict__ BT, int ldb,
            const float* __restrict__ bias,
            float* __restrict__ Cf, unsigned short* __restrict__ Cb,
            int M, int N, int K, int ldc, int nvalid, int dorelu) {
    int lane   = threadIdx.x & 31;
    int wid    = threadIdx.x >> 5;
    int mtile  = blockIdx.x;
    int ntbase = (blockIdx.y * (blockDim.x >> 5) + wid) * 4;   // first of 4 n-tiles
    if (ntbase * 16 >= N) return;
    int half = lane >> 4;          // K-half select
    int r    = lane & 15;          // A row within tile / B col within tile

    const unsigned short* ap = A + (size_t)(mtile * 16 + r) * lda;
    const unsigned short* bp0 = BT + (size_t)((ntbase + 0) * 16 + r) * ldb;
    const unsigned short* bp1 = BT + (size_t)((ntbase + 1) * 16 + r) * ldb;
    const unsigned short* bp2 = BT + (size_t)((ntbase + 2) * 16 + r) * ldb;
    const unsigned short* bp3 = BT + (size_t)((ntbase + 3) * 16 + r) * ldb;

    v8f acc0, acc1, acc2, acc3;
    #pragma unroll
    for (int i = 0; i < 8; ++i) { acc0[i] = 0.f; acc1[i] = 0.f; acc2[i] = 0.f; acc3[i] = 0.f; }

    for (int k0 = 0; k0 < K; k0 += 32) {
        FragU fa, fb0, fb1, fb2, fb3;
        // A 16x32 bf16 layout: lane<16 -> K {k0..k0+15}; lane>=16 -> +16
        fa.q[0] = *(const U4*)(ap + k0 + 16 * half);
        fa.q[1] = *(const U4*)(ap + k0 + 16 * half + 8);
        // B 32x16 bf16 (from transposed buffer): lane<16 -> K {k0..k0+15}; else +16
        int bo = k0 + 16 * half;
        fb0.q[0] = *(const U4*)(bp0 + bo); fb0.q[1] = *(const U4*)(bp0 + bo + 8);
        fb1.q[0] = *(const U4*)(bp1 + bo); fb1.q[1] = *(const U4*)(bp1 + bo + 8);
        fb2.q[0] = *(const U4*)(bp2 + bo); fb2.q[1] = *(const U4*)(bp2 + bo + 8);
        fb3.q[0] = *(const U4*)(bp3 + bo); fb3.q[1] = *(const U4*)(bp3 + bo + 8);
        acc0 = __builtin_amdgcn_wmma_f32_16x16x32_bf16(false, fa.v, false, fb0.v, (short)0, acc0, false, false);
        acc1 = __builtin_amdgcn_wmma_f32_16x16x32_bf16(false, fa.v, false, fb1.v, (short)0, acc1, false, false);
        acc2 = __builtin_amdgcn_wmma_f32_16x16x32_bf16(false, fa.v, false, fb2.v, (short)0, acc2, false, false);
        acc3 = __builtin_amdgcn_wmma_f32_16x16x32_bf16(false, fa.v, false, fb3.v, (short)0, acc3, false, false);
    }

    #pragma unroll
    for (int nt = 0; nt < 4; ++nt) {
        const v8f& acc = (nt == 0) ? acc0 : (nt == 1) ? acc1 : (nt == 2) ? acc2 : acc3;
        int col = (ntbase + nt) * 16 + r;
        float bv = (bias != nullptr && col < nvalid) ? bias[col] : 0.0f;
        #pragma unroll
        for (int i = 0; i < 8; ++i) {
            float x = acc[i] + bv;
            if (dorelu) x = fmaxf(x, 0.0f);
            int row = mtile * 16 + i + 8 * half;
            if (col < nvalid) {
                if (Cb) Cb[(size_t)row * ldc + col] = f2bf(x);
                else    Cf[(size_t)row * ldc + col] = x;
            }
        }
    }
}

// ---------------------------------------------------------------------------
extern "C" void kernel_launch(void* const* d_in, const int* in_sizes, int n_in,
                              void* d_out, int out_size, void* d_ws, size_t ws_size,
                              hipStream_t stream) {
    (void)in_sizes; (void)n_in; (void)out_size; (void)ws_size;

    const float* vertices = (const float*)d_in[0];   // [8,2048,3]
    const float* onehot   = (const float*)d_in[1];   // [8,16]
    const float* d0       = (const float*)d_in[2];   // [3,32]
    const float* W[8];  const float* Bv[8]; const float* DD[8];
    for (int i = 1; i <= 7; ++i) {
        W[i]  = (const float*)d_in[3 * i];
        Bv[i] = (const float*)d_in[3 * i + 1];
        DD[i] = (const float*)d_in[3 * i + 2];
    }
    const float* cw1 = (const float*)d_in[24]; const float* cb1 = (const float*)d_in[25];
    const float* cw2 = (const float*)d_in[26]; const float* cb2 = (const float*)d_in[27];
    const float* cw3 = (const float*)d_in[28]; const float* cb3 = (const float*)d_in[29];

    const int bs = 8, V0 = 2048, V1 = 512, V2 = 128;
    const int M0 = bs * V0, M1 = bs * V1, M2 = bs * V2;

    // workspace carving
    size_t off = 0;
    auto alloc = [&](size_t bytes) -> void* {
        void* p = (char*)d_ws + off;
        off = (off + bytes + 255) & ~(size_t)255;
        return p;
    };
    float* dnorm = (float*)alloc(8 * 96 * sizeof(float));          // d0, dd1..dd7
    unsigned short* wT[8];
    for (int i = 1; i <= 7; ++i) wT[i] = (unsigned short*)alloc((size_t)64 * (32 * i) * 2);
    unsigned short* cw1b = (unsigned short*)alloc((size_t)1024 * 1440 * 2);
    unsigned short* cw2b = (unsigned short*)alloc((size_t)512 * 1024 * 2);
    unsigned short* cw3b = (unsigned short*)alloc((size_t)64 * 512 * 2);
    int* ni0 = (int*)alloc((size_t)M0 * 32 * 4);
    int* ni1 = (int*)alloc((size_t)M1 * 32 * 4);
    int* ni2 = (int*)alloc((size_t)M2 * 32 * 4);
    int* ip0 = (int*)alloc((size_t)M1 * 4 * 4);
    int* ip1 = (int*)alloc((size_t)M2 * 4 * 4);
    float* fmA = (float*)alloc((size_t)M0 * 96 * 4);
    float* fmB = (float*)alloc((size_t)M1 * 192 * 4);
    float* fmC = (float*)alloc((size_t)M2 * 256 * 4);
    float* v1  = (float*)alloc((size_t)M1 * 3 * 4);
    float* v2  = (float*)alloc((size_t)M2 * 3 * 4);
    float* fbuf = (float*)alloc((size_t)M0 * 64 * 4);              // conv f, reused
    unsigned short* abf   = (unsigned short*)alloc((size_t)M0 * 1024 * 2); // A slices / h1
    unsigned short* h2b   = (unsigned short*)alloc((size_t)M0 * 512 * 2);
    unsigned short* fuseb = (unsigned short*)alloc((size_t)M0 * 1440 * 2);
    int* n1b = (int*)alloc((size_t)M0 * 4);
    int* n2b = (int*)alloc((size_t)M0 * 4);
    float* fg = (float*)alloc((size_t)8 * 256 * 4);

    auto cdiv = [](long long a, long long b) { return (int)((a + b - 1) / b); };

    // ---- prep: normalize direction kernels, transpose/convert weights ----
    hipLaunchKernelGGL(k_normcols, dim3(1), dim3(32), 0, stream, d0, dnorm);
    for (int i = 1; i <= 7; ++i)
        hipLaunchKernelGGL(k_normcols, dim3(1), dim3(32), 0, stream, DD[i], dnorm + 96 * i);
    for (int i = 1; i <= 7; ++i) {
        int K = 32 * i;
        hipLaunchKernelGGL(k_tcvt, dim3(cdiv((long long)K * 64, 256)), dim3(256), 0, stream,
                           W[i], wT[i], K, 64);
    }
    hipLaunchKernelGGL(k_cvt, dim3(cdiv(1024LL * 1440, 256)), dim3(256), 0, stream,
                       cw1, 1424, 1024, cw1b, 1440, 1024, 1424);
    hipLaunchKernelGGL(k_cvt, dim3(cdiv(512LL * 1024, 256)), dim3(256), 0, stream,
                       cw2, 1024, 512, cw2b, 1024, 512, 1024);
    hipLaunchKernelGGL(k_cvt, dim3(cdiv(64LL * 512, 256)), dim3(256), 0, stream,
                       cw3, 512, 50, cw3b, 512, 64, 512);

    // ---- stage 0 ----
    hipLaunchKernelGGL(k_ballq, dim3(cdiv(M0, 256)), dim3(256), 0, stream,
                       vertices, V0, vertices, V0, V0, 0.25f * 0.25f, 32, ni0, bs);
    hipLaunchKernelGGL(k_conv_surface, dim3(M0 / 8), dim3(256), 0, stream,
                       vertices, V0, ni0, dnorm, fmA, 96);
    // conv1 (C=32), conv2 (C=64): N=64 -> one wave covers the strip
    for (int l = 1; l <= 2; ++l) {
        int C = 32 * l;
        hipLaunchKernelGGL(k_cvt, dim3(cdiv((long long)M0 * C, 256)), dim3(256), 0, stream,
                           fmA, 96, M0, abf, C, M0, C);
        hipLaunchKernelGGL(k_gemm, dim3(M0 / 16, 1), dim3(32), 0, stream,
                           abf, C, wT[l], C, Bv[l], fbuf, (unsigned short*)nullptr,
                           M0, 64, C, 64, 64, 0);
        hipLaunchKernelGGL(k_conv_aggregate, dim3(M0 / 8), dim3(256), 0, stream,
                           vertices, V0, ni0, dnorm + 96 * l, fbuf, fmA, 96, 32 * l);
    }
    // pool1 (radius 0.25, nn=4), keep first V1 vertices
    hipLaunchKernelGGL(k_ballq, dim3(cdiv(M1, 256)), dim3(256), 0, stream,
                       vertices, V0, vertices, V1, V0, 0.25f * 0.25f, 4, ip0, bs);
    hipLaunchKernelGGL(k_pool, dim3(M1), dim3(256), 0, stream,
                       fmA, 96, 96, V0, ip0, vertices, fmB, 192, v1, V1);

    // ---- stage 1 ----
    hipLaunchKernelGGL(k_ballq, dim3(cdiv(M1, 256)), dim3(256), 0, stream,
                       v1, V1, v1, V1, V1, 0.39f * 0.39f, 32, ni1, bs);
    for (int l = 3; l <= 5; ++l) {
        int C = 32 * l;
        hipLaunchKernelGGL(k_cvt, dim3(cdiv((long long)M1 * C, 256)), dim3(256), 0, stream,
                           fmB, 192, M1, abf, C, M1, C);
        hipLaunchKernelGGL(k_gemm, dim3(M1 / 16, 1), dim3(32), 0, stream,
                           abf, C, wT[l], C, Bv[l], fbuf, (unsigned short*)nullptr,
                           M1, 64, C, 64, 64, 0);
        hipLaunchKernelGGL(k_conv_aggregate, dim3(M1 / 8), dim3(256), 0, stream,
                           v1, V1, ni1, dnorm + 96 * l, fbuf, fmB, 192, 32 * l - 64);
    }
    // pool2 (radius 0.39, nn=4)
    hipLaunchKernelGGL(k_ballq, dim3(cdiv(M2, 256)), dim3(256), 0, stream,
                       v1, V1, v1, V2, V1, 0.39f * 0.39f, 4, ip1, bs);
    hipLaunchKernelGGL(k_pool, dim3(M2), dim3(256), 0, stream,
                       fmB, 192, 192, V1, ip1, v1, fmC, 256, v2, V2);

    // ---- stage 2 ----
    hipLaunchKernelGGL(k_ballq, dim3(cdiv(M2, 256)), dim3(256), 0, stream,
                       v2, V2, v2, V2, V2, 0.63f * 0.63f, 32, ni2, bs);
    for (int l = 6; l <= 7; ++l) {
        int C = 32 * l;
        hipLaunchKernelGGL(k_cvt, dim3(cdiv((long long)M2 * C, 256)), dim3(256), 0, stream,
                           fmC, 256, M2, abf, C, M2, C);
        hipLaunchKernelGGL(k_gemm, dim3(M2 / 16, 1), dim3(32), 0, stream,
                           abf, C, wT[l], C, Bv[l], fbuf, (unsigned short*)nullptr,
                           M2, 64, C, 64, 64, 0);
        hipLaunchKernelGGL(k_conv_aggregate, dim3(M2 / 8), dim3(256), 0, stream,
                           v2, V2, ni2, dnorm + 96 * l, fbuf, fmC, 256, 32 * l - 160);
    }

    // ---- fusion ----
    hipLaunchKernelGGL(k_fglobal, dim3(8), dim3(256), 0, stream, fmC, fg);
    hipLaunchKernelGGL(k_nearest, dim3(cdiv(M0, 256)), dim3(256), 0, stream,
                       vertices, V0, v1, V1, n1b, bs);
    hipLaunchKernelGGL(k_nearest, dim3(cdiv(M0, 256)), dim3(256), 0, stream,
                       vertices, V0, v2, V2, n2b, bs);
    hipLaunchKernelGGL(k_fuse, dim3(M0), dim3(256), 0, stream,
                       fmA, fmB, fmC, n1b, n2b, fg, onehot, fuseb);

    // ---- MLP head (WMMA, bf16 intermediates) ----
    // h1 = relu(fuse @ cw1^T + cb1)  [16384, 1024] bf16
    // block of 4 waves covers 256 cols -> grid.y = 1024/256 = 4
    hipLaunchKernelGGL(k_gemm, dim3(M0 / 16, 4), dim3(128), 0, stream,
                       fuseb, 1440, cw1b, 1440, cb1, (float*)nullptr, abf,
                       M0, 1024, 1440, 1024, 1024, 1);
    // h2 = relu(h1 @ cw2^T + cb2)  [16384, 512] bf16 -> grid.y = 512/256 = 2
    hipLaunchKernelGGL(k_gemm, dim3(M0 / 16, 2), dim3(128), 0, stream,
                       abf, 1024, cw2b, 1024, cb2, (float*)nullptr, h2b,
                       M0, 512, 1024, 512, 512, 1);
    // out = h2 @ cw3^T + cb3  [16384, 50] fp32 (N padded to 64, guarded stores)
    hipLaunchKernelGGL(k_gemm, dim3(M0 / 16, 1), dim3(32), 0, stream,
                       h2b, 512, cw3b, 512, cb3, (float*)d_out, (unsigned short*)nullptr,
                       M0, 64, 512, 50, 50, 0);
}
